// WorldModelBackbone_85031762526793
// MI455X (gfx1250) — compile-verified, matrix operand order checked by reading, not run on hardware
//
#include <hip/hip_runtime.h>
#include <math.h>

// ---- model constants ----
#define BT      8          // B*T
#define S_LATN  256
#define S_TOTN  262        // 2 + REG + S_LAT
#define TOK     2096       // BT * S_TOT  (= 131 * 16)
#define Dm      1024
#define D_INn   768
#define DFFn    4096
#define Hn      16
#define HDn     64
#define DEPTHn  12
#define T_FR    8
#define CTXn    10
#define FREQn   256
#define CAPc    50.0f
#define QK_EPSc 1e-6f
#define RMS_EPSc 1e-6f
#define ROPE_BASEc 10000.0f
#define SIG_SCALEc 1000.0f
#define MAXPc   10000.0f

typedef __bf16 v16bf __attribute__((ext_vector_type(16)));
typedef float  v8f   __attribute__((ext_vector_type(8)));

#define GEMM_WAVES 4

__device__ __forceinline__ unsigned int pack_bf16(float lo, float hi)
{
    return (unsigned int)__builtin_bit_cast(unsigned short, (__bf16)lo) |
           ((unsigned int)__builtin_bit_cast(unsigned short, (__bf16)hi) << 16);
}

// ------------------------------------------------------------------
// Generic GEMM: C[M,N] = A[M,K] @ W[K,N] (+bias[N] | +residual[M,N])
// f32 in/out, bf16 WMMA math, f32 accumulate.
// Block tile: 64(M) x 64(N). The block's 4 waves SHARE one cooperatively
// staged 32x64 W tile per k-chunk (4 KB LDS); each wave owns one 16-row
// m-tile and runs 4 WMMAs per chunk (A-frag reused across the N strips).
// All 4 B fragments are loaded before the first WMMA so the 8 ds_load_b128
// form one clause and overlap the matrix pipe via partial s_wait_dscnt.
// W tile is staged into LDS in *fragment order*: for column-group nt,
// lane l's 16 bf16 operand values are contiguous -> 2x ds_load_b128.
//   (k,n) -> lane = 16*((k>>3)&1) + (n&15) ; j = (k&7) + 8*(k>>4)
// EPI: 0 = plain, 1 = +bias, 2 = +residual  (compile-time epilogue)
// Requires: M % 16 == 0, N % 64 == 0, K % 32 == 0.
// ------------------------------------------------------------------
template <int EPI>
__global__ __launch_bounds__(32 * GEMM_WAVES)
void gemm_wmma(const float* __restrict__ A, const float* __restrict__ W,
               float* __restrict__ C, const float* __restrict__ bias,
               const float* __restrict__ residual, int M, int N, int K)
{
    const int lane   = threadIdx.x & 31;
    const int wave   = threadIdx.x >> 5;
    const int tilesN  = N >> 6;                        // 64-column strips
    const int bm = blockIdx.x / tilesN;                // m-group of this block
    const int tn = blockIdx.x - bm * tilesN;           // n-strip of this block
    const int tm = bm * GEMM_WAVES + wave;             // this wave's m-tile
    const bool active = (tm < (M >> 4));
    const int m0 = tm << 4, n0 = tn << 6;
    const int ln15 = lane & 15;
    const int kb   = (lane >> 4) << 3;                 // per-lane K base: 0 or 8

    // block-shared staging tile in fragment order: 4 nt-groups x 32 lanes x 8 dwords
    __shared__ alignas(32) unsigned int lds_b[4 * 32 * 8];

    const float* arow = A + (size_t)(m0 + ln15) * K;

    v8f acc[4] = {};

    for (int k0 = 0; k0 < K; k0 += 32) {
        // cooperative stage of W[k0..k0+31, n0..n0+63] by all 128 threads:
        // each thread covers (k,k+1) x (n,n+1)
        #pragma unroll
        for (int it = 0; it < 4; ++it) {
            const int p = (it << 7) + threadIdx.x;     // 0..511
            const int n = (p & 31) << 1;               // even column 0..62
            const int k = (p >> 5) << 1;               // even row    0..30
            const float2 wa = *(const float2*)(W + (size_t)(k0 + k)     * N + n0 + n);
            const float2 wb = *(const float2*)(W + (size_t)(k0 + k + 1) * N + n0 + n);
            const int nt  = n >> 4;
            const int j2  = ((k & 7) + ((k >> 4) << 3)) >> 1;
            const int lt0 = (((k >> 3) & 1) << 4) + (n & 15);
            lds_b[(nt << 8) + ( lt0      << 3) + j2] = pack_bf16(wa.x, wb.x);
            lds_b[(nt << 8) + ((lt0 + 1) << 3) + j2] = pack_bf16(wa.y, wb.y);
        }
        if (wave == 0 && k0 + 32 < K)                  // spread prefetch of next k-slice
            __builtin_prefetch(W + (size_t)(k0 + 32 + lane) * N + n0, 0, 1);
        __syncthreads();

        if (active) {
            // A fragment: lane = row, K map {kb..kb+7, 16+kb..23+kb} (ISA 7.12.2)
            const float4 a0 = *(const float4*)(arow + k0 + kb);
            const float4 a1 = *(const float4*)(arow + k0 + kb + 4);
            const float4 a2 = *(const float4*)(arow + k0 + 16 + kb);
            const float4 a3 = *(const float4*)(arow + k0 + 16 + kb + 4);
            v16bf af;
            af[0]  = (__bf16)a0.x; af[1]  = (__bf16)a0.y;
            af[2]  = (__bf16)a0.z; af[3]  = (__bf16)a0.w;
            af[4]  = (__bf16)a1.x; af[5]  = (__bf16)a1.y;
            af[6]  = (__bf16)a1.z; af[7]  = (__bf16)a1.w;
            af[8]  = (__bf16)a2.x; af[9]  = (__bf16)a2.y;
            af[10] = (__bf16)a2.z; af[11] = (__bf16)a2.w;
            af[12] = (__bf16)a3.x; af[13] = (__bf16)a3.y;
            af[14] = (__bf16)a3.z; af[15] = (__bf16)a3.w;

            // load ALL four B fragments first (one ds clause, partial waits),
            // then issue four independent WMMAs back-to-back
            const unsigned int* lb = lds_b + (lane << 3);
            const v16bf bf0 = *(const v16bf*)(lb);
            const v16bf bf1 = *(const v16bf*)(lb + 256);
            const v16bf bf2 = *(const v16bf*)(lb + 512);
            const v16bf bf3 = *(const v16bf*)(lb + 768);
            acc[0] = __builtin_amdgcn_wmma_f32_16x16x32_bf16(
                         false, af, false, bf0, (short)0, acc[0], false, false);
            acc[1] = __builtin_amdgcn_wmma_f32_16x16x32_bf16(
                         false, af, false, bf1, (short)0, acc[1], false, false);
            acc[2] = __builtin_amdgcn_wmma_f32_16x16x32_bf16(
                         false, af, false, bf2, (short)0, acc[2], false, false);
            acc[3] = __builtin_amdgcn_wmma_f32_16x16x32_bf16(
                         false, af, false, bf3, (short)0, acc[3], false, false);
        }
        __syncthreads();
    }

    if (active) {
        #pragma unroll
        for (int nt = 0; nt < 4; ++nt) {
            const int cn = n0 + (nt << 4) + ln15;
            #pragma unroll
            for (int r = 0; r < 8; ++r) {
                const int cm = m0 + r + kb;    // C layout: M = r + 8*(lane>=16)
                float v = acc[nt][r];
                if (EPI == 1) v += bias[cn];
                if (EPI == 2) v += residual[(size_t)cm * N + cn];
                C[(size_t)cm * N + cn] = v;
            }
        }
    }
}

// ------------------------------------------------------------------
// RMSNorm over D=1024: y = x * rsqrt(mean(x^2)+eps) * w
// ------------------------------------------------------------------
__global__ __launch_bounds__(256)
void rmsnorm_kernel(const float* __restrict__ x, const float* __restrict__ w,
                    float* __restrict__ y)
{
    const int row = blockIdx.x;
    const int tid = threadIdx.x;
    __shared__ float red[256];
    const float* xr = x + (size_t)row * Dm;
    float s = 0.f;
    #pragma unroll
    for (int i = tid; i < Dm; i += 256) { float v = xr[i]; s += v * v; }
    red[tid] = s; __syncthreads();
    for (int off = 128; off > 0; off >>= 1) {
        if (tid < off) red[tid] += red[tid + off];
        __syncthreads();
    }
    const float inv = rsqrtf(red[0] / (float)Dm + RMS_EPSc);
    #pragma unroll
    for (int i = tid; i < Dm; i += 256)
        y[(size_t)row * Dm + i] = xr[i] * inv * w[i];
}

// ------------------------------------------------------------------
// Per-head QK RMSNorm + RoPE (in place). grid = (TOK, H), block = 64
// pos = s for spatial layers, t for temporal layers.
// ------------------------------------------------------------------
__global__ __launch_bounds__(64)
void qk_prep(float* __restrict__ q, float* __restrict__ k,
             const float* __restrict__ qn, const float* __restrict__ kn,
             int temporal)
{
    const int token = blockIdx.x;
    const int h   = blockIdx.y;
    const int tid = threadIdx.x;                 // 0..63 == HD
    const int bt  = token / S_TOTN;
    const int s   = token - bt * S_TOTN;
    const int pos = temporal ? (bt % T_FR) : s;

    __shared__ float xs[HDn];
    __shared__ float red[HDn];

    const float invf = __expf(-logf(ROPE_BASEc) * (float)((tid & 31) * 2) / (float)HDn);
    const float fr = (float)pos * invf;
    const float c  = __cosf(fr), sn = __sinf(fr);

    for (int which = 0; which < 2; ++which) {
        float* p = (which ? k : q) + (size_t)token * Dm + h * HDn;
        const float* nw = which ? kn : qn;
        const float v = p[tid];
        red[tid] = v * v;
        __syncthreads();
        for (int off = 32; off > 0; off >>= 1) {
            if (tid < off) red[tid] += red[tid + off];
            __syncthreads();
        }
        const float inv = rsqrtf(red[0] / (float)HDn + QK_EPSc);
        const float xn = v * inv * nw[tid];
        xs[tid] = xn;
        __syncthreads();
        const float rot = (tid < 32) ? -xs[tid + 32] : xs[tid - 32];
        p[tid] = xn * c + rot * sn;
        __syncthreads();
    }
}

// ------------------------------------------------------------------
// Spatial attention: one block (64 thr) per (bt, h, sq). S = 262 keys.
// logits -> tanh softcap -> softmax -> P@V (1 output dim per thread).
// ------------------------------------------------------------------
__global__ __launch_bounds__(64)
void attn_spatial(const float* __restrict__ q, const float* __restrict__ k,
                  const float* __restrict__ v, float* __restrict__ o)
{
    int id = blockIdx.x;
    const int sq = id % S_TOTN; id /= S_TOTN;
    const int h  = id % Hn;     id /= Hn;
    const int bt = id;
    const int tid = threadIdx.x;

    __shared__ float qrow[HDn];
    __shared__ float logits[S_TOTN];
    __shared__ float red[64];

    const size_t base = (size_t)bt * S_TOTN;
    qrow[tid] = q[(base + sq) * Dm + h * HDn + tid];
    __syncthreads();

    const float scale = rsqrtf((float)HDn);
    for (int j = tid; j < S_TOTN; j += 64) {
        const float* kp = k + (base + j) * Dm + h * HDn;
        float s = 0.f;
        #pragma unroll 8
        for (int d = 0; d < HDn; ++d) s += qrow[d] * kp[d];
        logits[j] = CAPc * tanhf(s * scale / CAPc);
    }
    __syncthreads();

    float m = -1e30f;
    for (int j = tid; j < S_TOTN; j += 64) m = fmaxf(m, logits[j]);
    red[tid] = m; __syncthreads();
    for (int off = 32; off > 0; off >>= 1) {
        if (tid < off) red[tid] = fmaxf(red[tid], red[tid + off]);
        __syncthreads();
    }
    const float mx = red[0];
    __syncthreads();

    float ps = 0.f;
    for (int j = tid; j < S_TOTN; j += 64) {
        const float e = __expf(logits[j] - mx);
        logits[j] = e; ps += e;
    }
    red[tid] = ps; __syncthreads();
    for (int off = 32; off > 0; off >>= 1) {
        if (tid < off) red[tid] += red[tid + off];
        __syncthreads();
    }
    const float inv = 1.f / red[0];

    float acc = 0.f;
    for (int j = 0; j < S_TOTN; ++j)
        acc += logits[j] * v[(base + j) * Dm + h * HDn + tid];
    o[(base + sq) * Dm + h * HDn + tid] = acc * inv;
}

// ------------------------------------------------------------------
// Temporal attention: one block (64 thr) per (s, h, tq); 8 keys along T.
// Mask: causal + CTX window + independent-frame self-only.
// ------------------------------------------------------------------
__global__ __launch_bounds__(64)
void attn_temporal(const float* __restrict__ q, const float* __restrict__ k,
                   const float* __restrict__ v, float* __restrict__ o,
                   const int* __restrict__ indep)
{
    int id = blockIdx.x;
    const int tq = id % T_FR; id /= T_FR;
    const int h  = id % Hn;   id /= Hn;
    const int s  = id;                    // 0..S_TOT-1 (B == 1)
    const int tid = threadIdx.x;

    __shared__ float qrow[HDn];
    __shared__ float logits[T_FR];

    const size_t qtok = (size_t)tq * S_TOTN + s;
    qrow[tid] = q[qtok * Dm + h * HDn + tid];
    __syncthreads();

    const bool ind = (indep[tq] != 0);
    if (tid < T_FR) {
        const int tk = tid;
        const bool ok = (tk <= tq) && ((tq - tk) <= CTXn) && (!ind || tk == tq);
        float val = -1e30f;
        if (ok) {
            const float* kp = k + ((size_t)tk * S_TOTN + s) * Dm + h * HDn;
            float sum = 0.f;
            #pragma unroll 8
            for (int d = 0; d < HDn; ++d) sum += qrow[d] * kp[d];
            val = CAPc * tanhf(sum * rsqrtf((float)HDn) / CAPc);
        }
        logits[tid] = val;
    }
    __syncthreads();

    float mx = -1e30f;
    #pragma unroll
    for (int j = 0; j < T_FR; ++j) mx = fmaxf(mx, logits[j]);
    float p[T_FR]; float ssum = 0.f;
    #pragma unroll
    for (int j = 0; j < T_FR; ++j) {
        p[j] = (logits[j] > -1e29f) ? __expf(logits[j] - mx) : 0.f;
        ssum += p[j];
    }
    const float inv = 1.f / ssum;
    float acc = 0.f;
    #pragma unroll
    for (int j = 0; j < T_FR; ++j)
        acc += p[j] * v[((size_t)j * S_TOTN + s) * Dm + h * HDn + tid];
    o[qtok * Dm + h * HDn + tid] = acc * inv;
}

// ------------------------------------------------------------------
// SwiGLU gate: u = silu(u) * g  (in place on u)
// ------------------------------------------------------------------
__global__ void silu_gate(float* __restrict__ u, const float* __restrict__ g, size_t n)
{
    const size_t i = (size_t)blockIdx.x * blockDim.x + threadIdx.x;
    if (i < n) {
        const float a = u[i];
        u[i] = (a / (1.f + __expf(-a))) * g[i];
    }
}

// ------------------------------------------------------------------
// Signal (timestep) embedding: sinusoidal -> silu MLP. grid = BT.
// ------------------------------------------------------------------
__global__ __launch_bounds__(256)
void sig_embed(const float* __restrict__ sl, const float* __restrict__ w1,
               const float* __restrict__ b1, const float* __restrict__ w2,
               const float* __restrict__ b2, float* __restrict__ sig)
{
    const int bt = blockIdx.x;
    const int tid = threadIdx.x;
    __shared__ float emb[FREQn];
    __shared__ float h1[Dm];
    const float tt = sl[bt] * SIG_SCALEc;
    const int half = FREQn / 2;
    if (tid < half) {
        const float f = __expf(-logf(MAXPc) * (float)tid / (float)half);
        emb[tid]        = __cosf(tt * f);
        emb[tid + half] = __sinf(tt * f);
    }
    __syncthreads();
    for (int d = tid; d < Dm; d += 256) {
        float a = b1[d];
        for (int f = 0; f < FREQn; ++f) a += emb[f] * w1[(size_t)f * Dm + d];
        h1[d] = a / (1.f + __expf(-a));     // silu
    }
    __syncthreads();
    for (int d = tid; d < Dm; d += 256) {
        float a = b2[d];
        for (int j = 0; j < Dm; ++j) a += h1[j] * w2[(size_t)j * Dm + d];
        sig[(size_t)bt * Dm + d] = a;
    }
}

// ------------------------------------------------------------------
// Action embedding with use_actions gating. grid = BT.
// ------------------------------------------------------------------
__global__ __launch_bounds__(256)
void act_embed(const float* __restrict__ actions, const int* __restrict__ use,
               const float* __restrict__ base, const float* __restrict__ aw,
               const float* __restrict__ ab, float* __restrict__ act)
{
    const int bt = blockIdx.x;
    const bool on = (use[bt] != 0);
    for (int d = threadIdx.x; d < Dm; d += 256) {
        float v = base[d];
        if (on) {
            float s = ab[d];
            #pragma unroll
            for (int a = 0; a < 8; ++a)
                s += actions[bt * 8 + a] * aw[(size_t)a * Dm + d];
            v += s;
        }
        act[(size_t)bt * Dm + d] = v;
    }
}

// ------------------------------------------------------------------
// Assemble token sequence: [sig, act, reg*4, lat*256] per frame.
// ------------------------------------------------------------------
__global__ void assemble(const float* __restrict__ sig, const float* __restrict__ act,
                         const float* __restrict__ regs, const float* __restrict__ lat,
                         float* __restrict__ x)
{
    const size_t i = (size_t)blockIdx.x * 256 + threadIdx.x;
    if (i >= (size_t)TOK * Dm) return;
    const int token = (int)(i / Dm);
    const int d = (int)(i - (size_t)token * Dm);
    const int bt = token / S_TOTN;
    const int s  = token - bt * S_TOTN;
    float v;
    if      (s == 0) v = sig[(size_t)bt * Dm + d];
    else if (s == 1) v = act[(size_t)bt * Dm + d];
    else if (s < 6)  v = regs[(size_t)(s - 2) * Dm + d];
    else             v = lat[((size_t)bt * S_LATN + (s - 6)) * Dm + d];
    x[i] = v;
}

// ------------------------------------------------------------------
// Final RMS on latent tokens only -> compact [BT*S_LAT, D]
// ------------------------------------------------------------------
__global__ __launch_bounds__(256)
void final_rms(const float* __restrict__ x, const float* __restrict__ w,
               float* __restrict__ y)
{
    const int r = blockIdx.x;                  // 0..BT*S_LAT-1
    const int bt = r >> 8, sl = r & 255;
    const int tid = threadIdx.x;
    __shared__ float red[256];
    const float* xr = x + ((size_t)bt * S_TOTN + 6 + sl) * Dm;
    float s = 0.f;
    #pragma unroll
    for (int i = tid; i < Dm; i += 256) { float v = xr[i]; s += v * v; }
    red[tid] = s; __syncthreads();
    for (int off = 128; off > 0; off >>= 1) {
        if (tid < off) red[tid] += red[tid + off];
        __syncthreads();
    }
    const float inv = rsqrtf(red[0] / (float)Dm + RMS_EPSc);
    #pragma unroll
    for (int i = tid; i < Dm; i += 256)
        y[(size_t)r * Dm + i] = xr[i] * inv * w[i];
}

// ------------------------------------------------------------------
extern "C" void kernel_launch(void* const* d_in, const int* in_sizes, int n_in,
                              void* d_out, int out_size, void* d_ws, size_t ws_size,
                              hipStream_t stream)
{
    const float* noisy   = (const float*)d_in[0];
    const float* sigl    = (const float*)d_in[1];
    const float* actions = (const float*)d_in[2];
    const int*   use_act = (const int*)  d_in[3];
    const int*   indep   = (const int*)  d_in[4];
    const float* in_w    = (const float*)d_in[5];
    const float* sw1     = (const float*)d_in[6];
    const float* sb1     = (const float*)d_in[7];
    const float* sw2     = (const float*)d_in[8];
    const float* sb2     = (const float*)d_in[9];
    const float* base_a  = (const float*)d_in[10];
    const float* act_w   = (const float*)d_in[11];
    const float* act_b   = (const float*)d_in[12];
    const float* regs    = (const float*)d_in[13];
    const float* ln1     = (const float*)d_in[14];
    const float* ln2     = (const float*)d_in[15];
    const float* wq      = (const float*)d_in[16];
    const float* wk      = (const float*)d_in[17];
    const float* wv      = (const float*)d_in[18];
    const float* wo      = (const float*)d_in[19];
    const float* qn      = (const float*)d_in[20];
    const float* kn      = (const float*)d_in[21];
    const float* w1      = (const float*)d_in[22];
    const float* w2      = (const float*)d_in[23];
    const float* w3      = (const float*)d_in[24];
    const float* fin_w   = (const float*)d_in[25];
    const float* out_w   = (const float*)d_in[26];
    const float* out_b   = (const float*)d_in[27];
    float* out = (float*)d_out;

    float* ws = (float*)d_ws;
    size_t off = 0;
    auto alloc = [&](size_t n) { float* p = ws + off; off += n; return p; };
    float* h   = alloc((size_t)TOK * Dm);
    float* g   = alloc((size_t)TOK * Dm);
    float* qb  = alloc((size_t)TOK * Dm);     // also reused as lat_in pre-assembly
    float* kb  = alloc((size_t)TOK * Dm);
    float* vb  = alloc((size_t)TOK * Dm);
    float* ob  = alloc((size_t)TOK * Dm);
    float* u   = alloc((size_t)TOK * DFFn);
    float* v2  = alloc((size_t)TOK * DFFn);
    float* sigbuf = alloc((size_t)BT * Dm);
    float* actbuf = alloc((size_t)BT * Dm);
    (void)in_sizes; (void)n_in; (void)out_size; (void)ws_size;

    auto gemm = [&](const float* A, const float* W, float* C, const float* bias,
                    const float* res, int M, int N, int K) {
        const int mgroups = (M / 16 + GEMM_WAVES - 1) / GEMM_WAVES;
        const int blocks  = mgroups * (N / 64);
        if (res)
            gemm_wmma<2><<<blocks, 32 * GEMM_WAVES, 0, stream>>>(A, W, C, nullptr, res, M, N, K);
        else if (bias)
            gemm_wmma<1><<<blocks, 32 * GEMM_WAVES, 0, stream>>>(A, W, C, bias, nullptr, M, N, K);
        else
            gemm_wmma<0><<<blocks, 32 * GEMM_WAVES, 0, stream>>>(A, W, C, nullptr, nullptr, M, N, K);
    };

    // ---- embeddings & token assembly ----
    gemm(noisy, in_w, qb, nullptr, nullptr, BT * S_LATN, Dm, D_INn);  // latents
    sig_embed<<<BT, 256, 0, stream>>>(sigl, sw1, sb1, sw2, sb2, sigbuf);
    act_embed<<<BT, 256, 0, stream>>>(actions, use_act, base_a, act_w, act_b, actbuf);
    {
        const size_t n = (size_t)TOK * Dm;
        assemble<<<(unsigned)((n + 255) / 256), 256, 0, stream>>>(sigbuf, actbuf, regs, qb, h);
    }

    // ---- transformer blocks ----
    for (int i = 0; i < DEPTHn; ++i) {
        const int temporal = (i % 4 == 0 && i != 0 && i != DEPTHn - 1) ? 1 : 0;

        rmsnorm_kernel<<<TOK, 256, 0, stream>>>(h, ln1 + (size_t)i * Dm, g);
        gemm(g, wq + (size_t)i * Dm * Dm, qb, nullptr, nullptr, TOK, Dm, Dm);
        gemm(g, wk + (size_t)i * Dm * Dm, kb, nullptr, nullptr, TOK, Dm, Dm);
        gemm(g, wv + (size_t)i * Dm * Dm, vb, nullptr, nullptr, TOK, Dm, Dm);
        qk_prep<<<dim3(TOK, Hn), 64, 0, stream>>>(qb, kb, qn + (size_t)i * HDn,
                                                  kn + (size_t)i * HDn, temporal);
        if (temporal)
            attn_temporal<<<S_TOTN * Hn * T_FR, 64, 0, stream>>>(qb, kb, vb, ob, indep);
        else
            attn_spatial<<<BT * Hn * S_TOTN, 64, 0, stream>>>(qb, kb, vb, ob);
        gemm(ob, wo + (size_t)i * Dm * Dm, h, nullptr, h, TOK, Dm, Dm);   // +residual

        rmsnorm_kernel<<<TOK, 256, 0, stream>>>(h, ln2 + (size_t)i * Dm, g);
        gemm(g, w1 + (size_t)i * Dm * DFFn, u,  nullptr, nullptr, TOK, DFFn, Dm);
        gemm(g, w2 + (size_t)i * Dm * DFFn, v2, nullptr, nullptr, TOK, DFFn, Dm);
        {
            const size_t n = (size_t)TOK * DFFn;
            silu_gate<<<(unsigned)((n + 255) / 256), 256, 0, stream>>>(u, v2, n);
        }
        gemm(u, w3 + (size_t)i * DFFn * Dm, h, nullptr, h, TOK, Dm, DFFn); // +residual
    }

    // ---- final norm + output projection ----
    final_rms<<<BT * S_LATN, 256, 0, stream>>>(h, fin_w, g);
    gemm(g, out_w, out, out_b, nullptr, BT * S_LATN, D_INn, Dm);
}
